// Graphormer_48232482735008
// MI455X (gfx1250) — matrix operand order repaired and checked by reading, not compiled, not run on hardware
//
#include <hip/hip_runtime.h>
#include <math.h>

#define B_  16
#define N_  128
#define NF_ 9
#define EF_ 3
#define MD_ 5
#define H_  24
#define D_  768
#define FF_ 3072
#define L_  12
#define O_  128
#define S_  129
#define SP_ 160
#define M_  (B_*S_)    // 2064
#define BH_ (B_*H_)    // 384

typedef __attribute__((ext_vector_type(16))) _Float16 v16h;
typedef __attribute__((ext_vector_type(8)))  _Float16 v8h;
typedef __attribute__((ext_vector_type(8)))  float    v8f;

union V16H { v16h v; v8h h[2]; };

// ---------------------------------------------------------------------------
// Graph attention bias: inner (i>=1, j>=1) part.
// One wave per (b,i,j). ei[5][24] built cooperatively in LDS, then 24 lanes
// contract against edge_dis_emb[:5] (120 MACs each).
// ---------------------------------------------------------------------------
__global__ __launch_bounds__(32) void gab_inner_kernel(
    const int* __restrict__ spatial_pos, const int* __restrict__ edge_input,
    const float* __restrict__ attn_bias, const float* __restrict__ edge_emb,
    const float* __restrict__ edge_dis_emb, const float* __restrict__ spatial_emb,
    float* __restrict__ gab)
{
    int i = blockIdx.x, j = blockIdx.y, b = blockIdx.z;
    int lane = threadIdx.x;
    __shared__ float ei[MD_*H_];
    const int* eidx = edge_input + ((((size_t)b*N_ + i)*N_ + j)*MD_)*EF_;
    for (int t = lane; t < MD_*H_; t += 32) {
        int d = t / H_, hh = t % H_;
        float s = 0.f;
        for (int e = 0; e < EF_; ++e)
            s += edge_emb[(size_t)eidx[d*EF_ + e]*H_ + hh];
        ei[t] = s * (1.0f/EF_);
    }
    __syncthreads();
    if (lane >= H_) return;
    int sp = spatial_pos[((size_t)b*N_ + i)*N_ + j];
    int spc = (sp == 0) ? 1 : sp;
    spc = (spc > 1) ? spc - 1 : spc;
    spc = (spc > MD_) ? MD_ : spc;
    float eo = 0.f;
    for (int d = 0; d < MD_; ++d)
        for (int hh = 0; hh < H_; ++hh)
            eo += ei[d*H_ + hh] * edge_dis_emb[((size_t)d*H_ + hh)*H_ + lane];
    float spb = spatial_emb[(size_t)sp*H_ + lane];
    float ab2 = 2.f * attn_bias[((size_t)b*S_ + (i+1))*S_ + (j+1)];
    gab[(((size_t)b*H_ + lane)*SP_ + (i+1))*SP_ + (j+1)] = ab2 + spb + eo / (float)spc;
}

// Row 0 (all k) and column 0 (q>=1): 2*attn_bias + virt_dist[h]
__global__ __launch_bounds__(256) void gab_border_kernel(
    const float* __restrict__ attn_bias, const float* __restrict__ vdist,
    float* __restrict__ gab)
{
    int t = blockIdx.x*256 + threadIdx.x;
    const int per = H_*(2*S_-1);
    if (t >= B_*per) return;
    int b = t / per; int r = t % per;
    int h = r / (2*S_-1); int p = r % (2*S_-1);
    int q, k;
    if (p < S_) { q = 0; k = p; } else { q = p - S_ + 1; k = 0; }
    gab[(((size_t)b*H_ + h)*SP_ + q)*SP_ + k] =
        2.f*attn_bias[((size_t)b*S_ + q)*S_ + k] + vdist[h];
}

// ---------------------------------------------------------------------------
// Node features -> h0 [B,S,D]
// ---------------------------------------------------------------------------
__global__ __launch_bounds__(256) void node_feat_kernel(
    const int* __restrict__ x, const int* __restrict__ ind, const int* __restrict__ outd,
    const float* __restrict__ atom_emb, const float* __restrict__ in_emb,
    const float* __restrict__ out_emb, const float* __restrict__ gtok,
    float* __restrict__ hbuf)
{
    size_t t = (size_t)blockIdx.x*256 + threadIdx.x;
    if (t >= (size_t)B_*S_*D_) return;
    int d = (int)(t % D_);
    int s = (int)((t / D_) % S_);
    int b = (int)(t / ((size_t)D_*S_));
    float v;
    if (s == 0) v = gtok[d];
    else {
        int n = s - 1;
        v = in_emb[(size_t)ind[b*N_ + n]*D_ + d] + out_emb[(size_t)outd[b*N_ + n]*D_ + d];
        const int* xr = x + ((size_t)b*N_ + n)*NF_;
        for (int f = 0; f < NF_; ++f) v += atom_emb[(size_t)xr[f]*D_ + d];
    }
    hbuf[t] = v;
}

// ---------------------------------------------------------------------------
// LayerNorm row -> f16
// ---------------------------------------------------------------------------
__global__ __launch_bounds__(256) void ln_cast_kernel(
    const float* __restrict__ X, const float* __restrict__ g,
    const float* __restrict__ bta, _Float16* __restrict__ Y, int Dd)
{
    int row = blockIdx.x, tid = threadIdx.x;
    const float* x = X + (size_t)row*Dd;
    __shared__ float sred[256];
    float s = 0.f;
    for (int d = tid; d < Dd; d += 256) s += x[d];
    sred[tid] = s; __syncthreads();
    for (int o = 128; o > 0; o >>= 1) { if (tid < o) sred[tid] += sred[tid+o]; __syncthreads(); }
    float mean = sred[0] / Dd; __syncthreads();
    float v = 0.f;
    for (int d = tid; d < Dd; d += 256) { float t = x[d]-mean; v += t*t; }
    sred[tid] = v; __syncthreads();
    for (int o = 128; o > 0; o >>= 1) { if (tid < o) sred[tid] += sred[tid+o]; __syncthreads(); }
    float rstd = rsqrtf(sred[0]/Dd + 1e-5f);
    for (int d = tid; d < Dd; d += 256)
        Y[(size_t)row*Dd + d] = (_Float16)((x[d]-mean)*rstd*g[d] + bta[d]);
}

// ---------------------------------------------------------------------------
// Weight cast+transpose: W[K,N] f32 -> Wt[N,K] f16
// ---------------------------------------------------------------------------
__global__ __launch_bounds__(256) void transpose_cast_kernel(
    const float* __restrict__ W, _Float16* __restrict__ Wt, int K, int Nn)
{
    size_t t = (size_t)blockIdx.x*256 + threadIdx.x;
    if (t >= (size_t)K*Nn) return;
    int n = (int)(t / K);
    int k = (int)(t % K);
    Wt[t] = (_Float16)W[(size_t)k*Nn + n];
}

// ---------------------------------------------------------------------------
// WMMA GEMM: C[M,N] = A[M,K](f16) * Bt[N,K]^T(f16) + bias, f32 out.
// Register-blocked: each wave computes 2 M-tiles x 4 N-tiles (32x64).
// 8 waves/block arranged 4(M) x 2(N): block covers 128x128 of C.
// Per K-step: 4 A-loads + 4 B-loads feed 8 independent WMMAs (24 B/WMMA/lane).
// ---------------------------------------------------------------------------
__global__ __launch_bounds__(256) void gemm_f16_kernel(
    const _Float16* __restrict__ A, const _Float16* __restrict__ Bt,
    const float* __restrict__ bias, float* __restrict__ C,
    int Mtiles, int Nn, int Kk)
{
    int wave = threadIdx.x >> 5;
    int lane = threadIdx.x & 31;
    int l15 = lane & 15, hi = lane >> 4;
    int wm = wave >> 1, wn = wave & 1;
    int tm0 = blockIdx.y*8 + wm*2;
    int tm1 = tm0 + 1;
    int tnb = blockIdx.x*8 + wn*4;
    if (tm0 >= Mtiles) return;                 // whole wave out of range
    int tm1c = (tm1 < Mtiles) ? tm1 : tm0;     // clamp loads for odd Mtiles

    const _Float16* ar0 = A + (size_t)(tm0*16 + l15)*Kk;
    const _Float16* ar1 = A + (size_t)(tm1c*16 + l15)*Kk;
    const _Float16* br0 = Bt + (size_t)((tnb+0)*16 + l15)*Kk;
    const _Float16* br1 = Bt + (size_t)((tnb+1)*16 + l15)*Kk;
    const _Float16* br2 = Bt + (size_t)((tnb+2)*16 + l15)*Kk;
    const _Float16* br3 = Bt + (size_t)((tnb+3)*16 + l15)*Kk;

    v8f c00 = {}, c01 = {}, c02 = {}, c03 = {};
    v8f c10 = {}, c11 = {}, c12 = {}, c13 = {};

    for (int kb = 0; kb < Kk; kb += 32) {
        V16H a0, a1, b0, b1, b2, b3;
        a0.h[0] = *(const v8h*)(ar0 + kb + 16*hi);
        a0.h[1] = *(const v8h*)(ar0 + kb + 16*hi + 8);
        a1.h[0] = *(const v8h*)(ar1 + kb + 16*hi);
        a1.h[1] = *(const v8h*)(ar1 + kb + 16*hi + 8);
        b0.v = *(const v16h*)(br0 + kb + 16*hi);
        b1.v = *(const v16h*)(br1 + kb + 16*hi);
        b2.v = *(const v16h*)(br2 + kb + 16*hi);
        b3.v = *(const v16h*)(br3 + kb + 16*hi);
        c00 = __builtin_amdgcn_wmma_f32_16x16x32_f16(false, a0.v, false, b0.v, (short)0, c00, false, false);
        c01 = __builtin_amdgcn_wmma_f32_16x16x32_f16(false, a0.v, false, b1.v, (short)0, c01, false, false);
        c02 = __builtin_amdgcn_wmma_f32_16x16x32_f16(false, a0.v, false, b2.v, (short)0, c02, false, false);
        c03 = __builtin_amdgcn_wmma_f32_16x16x32_f16(false, a0.v, false, b3.v, (short)0, c03, false, false);
        c10 = __builtin_amdgcn_wmma_f32_16x16x32_f16(false, a1.v, false, b0.v, (short)0, c10, false, false);
        c11 = __builtin_amdgcn_wmma_f32_16x16x32_f16(false, a1.v, false, b1.v, (short)0, c11, false, false);
        c12 = __builtin_amdgcn_wmma_f32_16x16x32_f16(false, a1.v, false, b2.v, (short)0, c12, false, false);
        c13 = __builtin_amdgcn_wmma_f32_16x16x32_f16(false, a1.v, false, b3.v, (short)0, c13, false, false);
    }

    v8f acc0[4] = { c00, c01, c02, c03 };
    v8f acc1[4] = { c10, c11, c12, c13 };
    #pragma unroll
    for (int j = 0; j < 4; ++j) {
        int col = (tnb + j)*16 + l15;
        float bv = bias ? bias[col] : 0.f;
        float* crow0 = C + (size_t)(tm0*16 + 8*hi) * Nn + col;
        #pragma unroll
        for (int r = 0; r < 8; ++r)
            crow0[(size_t)r * Nn] = acc0[j][r] + bv;
        if (tm1 < Mtiles) {
            float* crow1 = C + (size_t)(tm1*16 + 8*hi) * Nn + col;
            #pragma unroll
            for (int r = 0; r < 8; ++r)
                crow1[(size_t)r * Nn] = acc1[j][r] + bv;
        }
    }
}

// ---------------------------------------------------------------------------
// Repack projection output [B,S,D] -> per-head padded f16 buffers.
// transposed==0: dst[bh][s][32]; transposed==1: dst[bh][32][SP]
// ---------------------------------------------------------------------------
__global__ __launch_bounds__(256) void repack_kernel(
    const float* __restrict__ C, _Float16* __restrict__ dst,
    float scale, int transposed)
{
    int t = blockIdx.x*256 + threadIdx.x;
    if (t >= BH_*SP_*32) return;
    int d = t % 32;
    int s = (t / 32) % SP_;
    int bh = t / (32*SP_);
    int b = bh / H_, h = bh % H_;
    float val = 0.f;
    if (s < S_) val = C[((size_t)(b*S_ + s))*D_ + h*32 + d] * scale;
    if (!transposed) dst[(size_t)bh*SP_*32 + s*32 + d] = (_Float16)val;
    else             dst[(size_t)bh*SP_*32 + d*SP_ + s] = (_Float16)val;
}

// ---------------------------------------------------------------------------
// Attention scores: att[bh][q][k] = (Q K^T)[q][k] + gab[bh][q][k]
// ---------------------------------------------------------------------------
__global__ __launch_bounds__(32) void scores_kernel(
    const _Float16* __restrict__ q16, const _Float16* __restrict__ k16,
    const float* __restrict__ gab, float* __restrict__ att)
{
    int kt = blockIdx.x, qt = blockIdx.y, bh = blockIdx.z;
    int lane = threadIdx.x;
    int l15 = lane & 15, hi = lane >> 4;
    const _Float16* A  = q16 + (size_t)bh*SP_*32;
    const _Float16* Bt = k16 + (size_t)bh*SP_*32;
    V16H a, b;
    const _Float16* ap = A + (size_t)(qt*16 + l15)*32;
    a.h[0] = *(const v8h*)(ap + 16*hi);
    a.h[1] = *(const v8h*)(ap + 16*hi + 8);
    b.v    = *(const v16h*)(Bt + (size_t)(kt*16 + l15)*32 + 16*hi);
    v8f acc = {};
    acc = __builtin_amdgcn_wmma_f32_16x16x32_f16(false, a.v, false, b.v,
                                                 (short)0, acc, false, false);
    int col = kt*16 + l15;
    #pragma unroll
    for (int r = 0; r < 8; ++r) {
        int q = qt*16 + r + 8*hi;
        float gv = (q < S_ && col < S_) ? gab[((size_t)bh*SP_ + q)*SP_ + col] : 0.f;
        att[((size_t)bh*SP_ + q)*SP_ + col] = acc[r] + gv;
    }
}

// ---------------------------------------------------------------------------
// Softmax over k (valid k < S); pads with exact zeros. In place (f32 probs).
// ---------------------------------------------------------------------------
__global__ __launch_bounds__(256) void softmax_kernel(float* __restrict__ att)
{
    int q = blockIdx.x, bh = blockIdx.y, tid = threadIdx.x;
    float* row = att + (size_t)bh*SP_*SP_ + (size_t)q*SP_;
    if (q >= S_) {
        for (int k = tid; k < SP_; k += 256) row[k] = 0.f;
        return;
    }
    __shared__ float sred[256];
    float mx = -3.0e38f;
    for (int k = tid; k < S_; k += 256) mx = fmaxf(mx, row[k]);
    sred[tid] = mx; __syncthreads();
    for (int o = 128; o > 0; o >>= 1) { if (tid < o) sred[tid] = fmaxf(sred[tid], sred[tid+o]); __syncthreads(); }
    mx = sred[0]; __syncthreads();
    float sum = 0.f;
    for (int k = tid; k < S_; k += 256) { float e = expf(row[k]-mx); row[k] = e; sum += e; }
    sred[tid] = sum; __syncthreads();
    for (int o = 128; o > 0; o >>= 1) { if (tid < o) sred[tid] += sred[tid+o]; __syncthreads(); }
    float inv = 1.f / sred[0];
    for (int k = tid; k < SP_; k += 256) row[k] = (k < S_) ? row[k]*inv : 0.f;
}

// ---------------------------------------------------------------------------
// Context: ctx[b,q,h*32+d] = sum_k P[bh][q][k] * V[bh][k][d]
// A read f32 (probs), converted in-register; Vt is [bh][32][SP].
// ---------------------------------------------------------------------------
__global__ __launch_bounds__(32) void ctx_kernel(
    const float* __restrict__ att, const _Float16* __restrict__ vt16,
    float* __restrict__ ctx)
{
    int nt = blockIdx.x;   // 0..1 (dk tiles)
    int qt = blockIdx.y;   // 0..8 (covers q < 144 >= S)
    int bh = blockIdx.z;
    int lane = threadIdx.x;
    int l15 = lane & 15, hi = lane >> 4;
    const float*    P  = att  + (size_t)bh*SP_*SP_;
    const _Float16* Vt = vt16 + (size_t)bh*32*SP_;
    const float* prow = P + (size_t)(qt*16 + l15)*SP_;
    const _Float16* vrow = Vt + (size_t)(nt*16 + l15)*SP_;
    v8f acc = {};
    for (int kb = 0; kb < SP_; kb += 32) {
        V16H a, b;
        const float* pp = prow + kb + 16*hi;
        #pragma unroll
        for (int e = 0; e < 8; ++e) {
            a.v[e]   = (_Float16)pp[e];
            a.v[e+8] = (_Float16)pp[e+8];
        }
        b.v = *(const v16h*)(vrow + kb + 16*hi);
        acc = __builtin_amdgcn_wmma_f32_16x16x32_f16(false, a.v, false, b.v,
                                                     (short)0, acc, false, false);
    }
    int b = bh / H_, h = bh % H_;
    int d = nt*16 + l15;
    #pragma unroll
    for (int r = 0; r < 8; ++r) {
        int q = qt*16 + r + 8*hi;
        if (q < S_)
            ctx[((size_t)(b*S_ + q))*D_ + h*32 + d] = acc[r];
    }
}

// ---------------------------------------------------------------------------
// Elementwise helpers
// ---------------------------------------------------------------------------
__global__ __launch_bounds__(256) void cast_kernel(
    const float* __restrict__ src, _Float16* __restrict__ dst, size_t n)
{
    size_t t = (size_t)blockIdx.x*256 + threadIdx.x;
    if (t < n) dst[t] = (_Float16)src[t];
}

__global__ __launch_bounds__(256) void gelu_cast_kernel(
    const float* __restrict__ src, _Float16* __restrict__ dst, size_t n)
{
    size_t t = (size_t)blockIdx.x*256 + threadIdx.x;
    if (t < n) {
        float xv = src[t];
        dst[t] = (_Float16)(0.5f*xv*(1.f + erff(xv*0.70710678118654752f)));
    }
}

__global__ __launch_bounds__(256) void residual_kernel(
    float* __restrict__ hbuf, const float* __restrict__ src, size_t n)
{
    size_t t = (size_t)blockIdx.x*256 + threadIdx.x;
    if (t < n) hbuf[t] += src[t];
}

// ---------------------------------------------------------------------------
// Final LN on h[:,0] + head: out[b,o]
// ---------------------------------------------------------------------------
__global__ __launch_bounds__(256) void final_kernel(
    const float* __restrict__ hbuf, const float* __restrict__ fg,
    const float* __restrict__ fb, const float* __restrict__ ow,
    const float* __restrict__ ob, float* __restrict__ out)
{
    int b = blockIdx.x, tid = threadIdx.x;
    __shared__ float row[D_];
    __shared__ float sred[256];
    const float* x = hbuf + (size_t)b*S_*D_;   // row s=0
    float s = 0.f;
    for (int d = tid; d < D_; d += 256) s += x[d];
    sred[tid] = s; __syncthreads();
    for (int o = 128; o > 0; o >>= 1) { if (tid < o) sred[tid] += sred[tid+o]; __syncthreads(); }
    float mean = sred[0] / D_; __syncthreads();
    float v = 0.f;
    for (int d = tid; d < D_; d += 256) { float t = x[d]-mean; v += t*t; }
    sred[tid] = v; __syncthreads();
    for (int o = 128; o > 0; o >>= 1) { if (tid < o) sred[tid] += sred[tid+o]; __syncthreads(); }
    float rstd = rsqrtf(sred[0]/D_ + 1e-5f);
    for (int d = tid; d < D_; d += 256) row[d] = (x[d]-mean)*rstd*fg[d] + fb[d];
    __syncthreads();
    for (int o = tid; o < O_; o += 256) {
        float acc = ob[o];
        for (int d = 0; d < D_; ++d) acc += row[d]*ow[(size_t)d*O_ + o];
        out[(size_t)b*O_ + o] = acc;
    }
}

// ---------------------------------------------------------------------------
extern "C" void kernel_launch(void* const* d_in, const int* in_sizes, int n_in,
                              void* d_out, int out_size, void* d_ws, size_t ws_size,
                              hipStream_t stream)
{
    (void)in_sizes; (void)n_in; (void)out_size; (void)ws_size;
    const int*   x        = (const int*)  d_in[0];
    const int*   in_deg   = (const int*)  d_in[1];
    const int*   out_deg  = (const int*)  d_in[2];
    const int*   sp_pos   = (const int*)  d_in[3];
    const int*   edge_in  = (const int*)  d_in[4];
    const float* attn_b   = (const float*)d_in[5];
    const float* atom_emb = (const float*)d_in[6];
    const float* ind_emb  = (const float*)d_in[7];
    const float* outd_emb = (const float*)d_in[8];
    const float* gtok     = (const float*)d_in[9];
    const float* edge_emb = (const float*)d_in[10];
    const float* edis_emb = (const float*)d_in[11];
    const float* sp_emb   = (const float*)d_in[12];
    const float* vdist    = (const float*)d_in[13];
    const float* ln1_g    = (const float*)d_in[14];
    const float* ln1_b    = (const float*)d_in[15];
    const float* wq       = (const float*)d_in[16];
    const float* bq       = (const float*)d_in[17];
    const float* wk       = (const float*)d_in[18];
    const float* bk       = (const float*)d_in[19];
    const float* wv       = (const float*)d_in[20];
    const float* bv       = (const float*)d_in[21];
    const float* wo       = (const float*)d_in[22];
    const float* bo       = (const float*)d_in[23];
    const float* ln2_g    = (const float*)d_in[24];
    const float* ln2_b    = (const float*)d_in[25];
    const float* w1       = (const float*)d_in[26];
    const float* b1       = (const float*)d_in[27];
    const float* w2       = (const float*)d_in[28];
    const float* b2       = (const float*)d_in[29];
    const float* fg       = (const float*)d_in[30];
    const float* fb       = (const float*)d_in[31];
    const float* ow       = (const float*)d_in[32];
    const float* ob       = (const float*)d_in[33];
    float* out = (float*)d_out;

    char* base = (char*)d_ws;
    size_t off = 0;
    auto take = [&](size_t bytes) -> void* {
        void* p = base + off;
        off += (bytes + 255) & ~(size_t)255;
        return p;
    };
    float*    gab  = (float*)   take((size_t)BH_*SP_*SP_*sizeof(float));
    float*    att  = (float*)   take((size_t)BH_*SP_*SP_*sizeof(float));
    float*    hbuf = (float*)   take((size_t)M_*D_*sizeof(float));
    _Float16* y16  = (_Float16*)take((size_t)M_*D_*2);
    float*    cbuf = (float*)   take((size_t)M_*FF_*sizeof(float));
    _Float16* a16  = (_Float16*)take((size_t)M_*FF_*2);
    _Float16* q16  = (_Float16*)take((size_t)BH_*SP_*32*2);
    _Float16* k16  = (_Float16*)take((size_t)BH_*SP_*32*2);
    _Float16* vt16 = (_Float16*)take((size_t)BH_*SP_*32*2);
    _Float16* wt16 = (_Float16*)take((size_t)D_*FF_*2);

    // ---- graph attention bias + node features ----
    gab_inner_kernel<<<dim3(N_, N_, B_), 32, 0, stream>>>(
        sp_pos, edge_in, attn_b, edge_emb, edis_emb, sp_emb, gab);
    {
        int tot = B_*H_*(2*S_-1);
        gab_border_kernel<<<(tot+255)/256, 256, 0, stream>>>(attn_b, vdist, gab);
    }
    {
        size_t tot = (size_t)B_*S_*D_;
        node_feat_kernel<<<(unsigned)((tot+255)/256), 256, 0, stream>>>(
            x, in_deg, out_deg, atom_emb, ind_emb, outd_emb, gtok, hbuf);
    }

    const int Mtiles = M_/16;   // 129
    // Block covers 128 rows (8 M-tiles) x 128 cols (8 N-tiles).
    const dim3 gemmGridD(D_/128,  (Mtiles+7)/8);   // (6, 17)
    const dim3 gemmGridF(FF_/128, (Mtiles+7)/8);   // (24, 17)
    const float qscale = 0.17677669529663687f;  // 1/sqrt(32)
    const unsigned repackBlocks = (BH_*SP_*32 + 255)/256;
    const unsigned mdBlocks = (unsigned)(((size_t)M_*D_ + 255)/256);
    const unsigned mfBlocks = (unsigned)(((size_t)M_*FF_ + 255)/256);
    const unsigned ddBlocks = (unsigned)(((size_t)D_*D_ + 255)/256);
    const unsigned dfBlocks = (unsigned)(((size_t)D_*FF_ + 255)/256);

    for (int l = 0; l < L_; ++l) {
        // ---- attention ----
        ln_cast_kernel<<<M_, 256, 0, stream>>>(hbuf, ln1_g + l*D_, ln1_b + l*D_, y16, D_);

        transpose_cast_kernel<<<ddBlocks, 256, 0, stream>>>(wq + (size_t)l*D_*D_, wt16, D_, D_);
        gemm_f16_kernel<<<gemmGridD, 256, 0, stream>>>(y16, wt16, bq + l*D_, cbuf, Mtiles, D_, D_);
        repack_kernel<<<repackBlocks, 256, 0, stream>>>(cbuf, q16, qscale, 0);

        transpose_cast_kernel<<<ddBlocks, 256, 0, stream>>>(wk + (size_t)l*D_*D_, wt16, D_, D_);
        gemm_f16_kernel<<<gemmGridD, 256, 0, stream>>>(y16, wt16, bk + l*D_, cbuf, Mtiles, D_, D_);
        repack_kernel<<<repackBlocks, 256, 0, stream>>>(cbuf, k16, 1.0f, 0);

        transpose_cast_kernel<<<ddBlocks, 256, 0, stream>>>(wv + (size_t)l*D_*D_, wt16, D_, D_);
        gemm_f16_kernel<<<gemmGridD, 256, 0, stream>>>(y16, wt16, bv + l*D_, cbuf, Mtiles, D_, D_);
        repack_kernel<<<repackBlocks, 256, 0, stream>>>(cbuf, vt16, 1.0f, 1);

        scores_kernel<<<dim3(SP_/16, SP_/16, BH_), 32, 0, stream>>>(q16, k16, gab, att);
        softmax_kernel<<<dim3(SP_, BH_), 256, 0, stream>>>(att);
        ctx_kernel<<<dim3(2, 9, BH_), 32, 0, stream>>>(att, vt16, cbuf);

        cast_kernel<<<mdBlocks, 256, 0, stream>>>(cbuf, a16, (size_t)M_*D_);
        transpose_cast_kernel<<<ddBlocks, 256, 0, stream>>>(wo + (size_t)l*D_*D_, wt16, D_, D_);
        gemm_f16_kernel<<<gemmGridD, 256, 0, stream>>>(a16, wt16, bo + l*D_, cbuf, Mtiles, D_, D_);
        residual_kernel<<<mdBlocks, 256, 0, stream>>>(hbuf, cbuf, (size_t)M_*D_);

        // ---- FFN ----
        ln_cast_kernel<<<M_, 256, 0, stream>>>(hbuf, ln2_g + l*D_, ln2_b + l*D_, y16, D_);
        transpose_cast_kernel<<<dfBlocks, 256, 0, stream>>>(w1 + (size_t)l*D_*FF_, wt16, D_, FF_);
        gemm_f16_kernel<<<gemmGridF, 256, 0, stream>>>(y16, wt16, b1 + l*FF_, cbuf, Mtiles, FF_, D_);
        gelu_cast_kernel<<<mfBlocks, 256, 0, stream>>>(cbuf, a16, (size_t)M_*FF_);
        transpose_cast_kernel<<<dfBlocks, 256, 0, stream>>>(w2 + (size_t)l*FF_*D_, wt16, FF_, D_);
        gemm_f16_kernel<<<gemmGridD, 256, 0, stream>>>(a16, wt16, b2 + l*D_, cbuf, Mtiles, D_, FF_);
        residual_kernel<<<mdBlocks, 256, 0, stream>>>(hbuf, cbuf, (size_t)M_*D_);
    }

    final_kernel<<<B_, 256, 0, stream>>>(hbuf, fg, fb, ow, ob, out);
}
